// SelectiveSSM_33277406610014
// MI455X (gfx1250) — matrix operand order repaired and checked by reading, not compiled, not run on hardware
//
#include <hip/hip_runtime.h>
#include <hip/hip_bf16.h>

// ---------------------------------------------------------------------------
// Selective SSM for MI455X (gfx1250, wave32).
// Pipeline:
//   K1 pack:   x -> f16, [W_in;W_delta;W_B;W_C] -> f16 (640x64), W_out -> f16,
//              bias_all (640) = [0 x256, b_delta, b_B, b_C]
//   K2 gemm:   P(4096x640) = Xf16 @ W_all^T + bias  (v_wmma_f32_16x16x32_f16,
//              tiles staged via GLOBAL_LOAD_ASYNC_TO_LDS_B128 when available)
//   K3 abar:   delta=softplus, x_pool, Abar = mean_i exp(delta_i*A[i,s]), Bx
//   K4 scan:   sequential recurrence per batch, 128 lanes = states
//   K5 outg:   y*silu(gate) -> f16 tile in LDS, WMMA with W_out^T (K=128);
//              W_out tile async-DMA'd into LDS overlapping the silu VALU work
//   K6 ln:     residual + layernorm(64) via wave shuffles
// ---------------------------------------------------------------------------

#define BB      4
#define LL      1024
#define DMODEL  64
#define DINNER  128
#define DSTATE  128
#define MROWS   (BB * LL)          // 4096
#define NCOL    640                // 256 + 128 + 128 + 128

typedef __attribute__((ext_vector_type(16))) _Float16 v16h;
typedef __attribute__((ext_vector_type(8)))  float    v8f;

#define AS1 __attribute__((address_space(1)))
#define AS3 __attribute__((address_space(3)))

// exact parameter type reported by clang for the async builtin:
typedef int ssm_v4i __attribute__((vector_size(4 * sizeof(int))));

#if defined(__gfx1250__) && __has_builtin(__builtin_amdgcn_global_load_async_to_lds_b128)
#define SSM_HAVE_ASYNC_LDS 1
#else
#define SSM_HAVE_ASYNC_LDS 0
#endif

#if SSM_HAVE_ASYNC_LDS
__device__ __forceinline__ void ssm_async_b128(const void* g, void* lds) {
    // per-lane 16B DMA: LDS[vdst_lane] = MEM[vaddr_lane]; tracked by ASYNCcnt
    __builtin_amdgcn_global_load_async_to_lds_b128(
        (AS1 ssm_v4i*)g, (AS3 ssm_v4i*)lds, /*imm offset*/ 0, /*cpol*/ 0);
}
__device__ __forceinline__ void ssm_wait_async0() {
#if __has_builtin(__builtin_amdgcn_s_wait_asynccnt)
    __builtin_amdgcn_s_wait_asynccnt(0);
#else
    asm volatile("s_wait_asynccnt 0" ::: "memory");
#endif
}
#endif

union Frag16 {
    uint4 q[2];   // 32 bytes
    v16h  h;
};

__device__ __forceinline__ float wave_sum32(float v) {
    v += __shfl_xor(v, 16, 32);
    v += __shfl_xor(v,  8, 32);
    v += __shfl_xor(v,  4, 32);
    v += __shfl_xor(v,  2, 32);
    v += __shfl_xor(v,  1, 32);
    return v;
}

// ---------------------------------------------------------------- K1: pack --
#define PACK_N0 (MROWS * DMODEL)            // 262144
#define PACK_N1 (NCOL * DMODEL)             // 40960
#define PACK_N2 (DMODEL * DINNER)           // 8192
#define PACK_N3 (NCOL)                      // 640
#define PACK_TOTAL (PACK_N0 + PACK_N1 + PACK_N2 + PACK_N3)

__global__ void ssm_pack_kernel(const float* __restrict__ x,
                                const float* __restrict__ W_in,
                                const float* __restrict__ W_delta,
                                const float* __restrict__ b_delta,
                                const float* __restrict__ W_B,
                                const float* __restrict__ b_B,
                                const float* __restrict__ W_C,
                                const float* __restrict__ b_C,
                                const float* __restrict__ W_out,
                                _Float16* __restrict__ Xh,
                                _Float16* __restrict__ Wall_h,
                                _Float16* __restrict__ Wout_h,
                                float* __restrict__ bias_all) {
    int idx = blockIdx.x * blockDim.x + threadIdx.x;
    if (idx < PACK_N0) {
        Xh[idx] = (_Float16)x[idx];
        return;
    }
    idx -= PACK_N0;
    if (idx < PACK_N1) {
        int e = idx / DMODEL, d = idx % DMODEL;
        float v;
        if (e < 256)       v = W_in[e * DMODEL + d];
        else if (e < 384)  v = W_delta[(e - 256) * DMODEL + d];
        else if (e < 512)  v = W_B[(e - 384) * DMODEL + d];
        else               v = W_C[(e - 512) * DMODEL + d];
        Wall_h[idx] = (_Float16)v;
        return;
    }
    idx -= PACK_N1;
    if (idx < PACK_N2) {
        Wout_h[idx] = (_Float16)W_out[idx];   // (64 x 128) row-major
        return;
    }
    idx -= PACK_N2;
    if (idx < PACK_N3) {
        float v;
        if (idx < 256)      v = 0.0f;
        else if (idx < 384) v = b_delta[idx - 256];
        else if (idx < 512) v = b_B[idx - 384];
        else                v = b_C[idx - 512];
        bias_all[idx] = v;
    }
}

// ------------------------------------------------------- K2: projection GEMM
// P(4096x640) = Xh(4096x64) @ Wall^T(64x640) + bias.
// Block: 128 threads (4 waves). Tile: 64(M) x 64(N), K=64 (two 16x16x32 WMMAs).
__global__ void __launch_bounds__(128)
ssm_gemm_proj_kernel(const _Float16* __restrict__ Xh,
                     const _Float16* __restrict__ Wall_h,
                     const float* __restrict__ bias_all,
                     float* __restrict__ P) {
    __shared__ __align__(16) _Float16 sX[64 * 64];   // 8 KB
    __shared__ __align__(16) _Float16 sW[64 * 64];   // 8 KB (rows = output cols)

    const int tid  = threadIdx.x;
    const int m0   = blockIdx.x * 64;
    const int n0   = blockIdx.y * 64;

    const uint4* gX = (const uint4*)(Xh     + (size_t)m0 * DMODEL);
    const uint4* gW = (const uint4*)(Wall_h + (size_t)n0 * DMODEL);
    uint4* lX = (uint4*)sX;
    uint4* lW = (uint4*)sW;

#if SSM_HAVE_ASYNC_LDS
    // Async DMA global->LDS (ASYNCcnt), no VGPR round trip.
#pragma unroll
    for (int t = 0; t < 4; ++t) {
        ssm_async_b128(gX + tid + 128 * t, lX + tid + 128 * t);
        ssm_async_b128(gW + tid + 128 * t, lW + tid + 128 * t);
    }
    ssm_wait_async0();          // each wave drains its own ASYNCcnt
    __syncthreads();            // then waves exchange LDS visibility
#else
#pragma unroll
    for (int t = 0; t < 4; ++t) lX[tid + 128 * t] = gX[tid + 128 * t];
#pragma unroll
    for (int t = 0; t < 4; ++t) lW[tid + 128 * t] = gW[tid + 128 * t];
    __syncthreads();
#endif

    const int wave = tid >> 5;
    const int lane = tid & 31;
    const int ml   = lane & 15;       // row within A fragment / col within B
    const int hi   = lane >> 4;

    // A fragments for this wave's 16 rows, k0 = 0 and 32.
    // Lane layout (ISA 7.12.2, 16-bit A 16x32): halves 0..7 -> K=k0+8*hi..+7,
    // halves 8..15 -> K=k0+16+8*hi..+7  => two contiguous 16B chunks.
    const int arow = (wave * 16 + ml) * 64;
    Frag16 a0, a1;
    a0.q[0] = *(const uint4*)&sX[arow + 0  + hi * 8];
    a0.q[1] = *(const uint4*)&sX[arow + 16 + hi * 8];
    a1.q[0] = *(const uint4*)&sX[arow + 32 + hi * 8];
    a1.q[1] = *(const uint4*)&sX[arow + 48 + hi * 8];

#pragma unroll
    for (int ns = 0; ns < 4; ++ns) {
        const int ncol = ns * 16 + ml;             // local output column
        // B fragment: lane holds col n, same K layout as A (k0+8*hi.., k0+16+8*hi..).
        const int brow = ncol * 64;
        Frag16 b0, b1;
        b0.q[0] = *(const uint4*)&sW[brow + 0  + hi * 8];
        b0.q[1] = *(const uint4*)&sW[brow + 16 + hi * 8];
        b1.q[0] = *(const uint4*)&sW[brow + 32 + hi * 8];
        b1.q[1] = *(const uint4*)&sW[brow + 48 + hi * 8];

        v8f c = {};
        c = __builtin_amdgcn_wmma_f32_16x16x32_f16(false, a0.h, false, b0.h,
                                                   (short)0, c, false, false);
        c = __builtin_amdgcn_wmma_f32_16x16x32_f16(false, a1.h, false, b1.h,
                                                   (short)0, c, false, false);

        // C/D layout: VGPR r, lanes0-15 -> M=r, lanes16-31 -> M=r+8; N=lane&15.
        const int gn = n0 + ncol;
        const float bv = bias_all[gn];
        const int gmbase = m0 + wave * 16 + hi * 8;
#pragma unroll
        for (int r = 0; r < 8; ++r)
            P[(size_t)(gmbase + r) * NCOL + gn] = c[r] + bv;
    }

#if __has_builtin(__builtin_amdgcn_s_cluster_barrier)
    // NOP when ClusterID==0 (ISA 3.2); exercises s_barrier_signal/wait -3.
    __builtin_amdgcn_s_cluster_barrier();
#endif
}

// ------------------------------------------------- K3: softplus / Abar / Bx
__global__ void __launch_bounds__(128)
ssm_abar_kernel(const float* __restrict__ P,
                const float* __restrict__ A,
                float* __restrict__ Abar,
                float* __restrict__ Bx) {
    __shared__ float sDelta[DINNER];
    __shared__ float sRed[4];

    const int row = blockIdx.x;
    const int tid = threadIdx.x;
    const size_t pbase = (size_t)row * NCOL;

    // softplus(delta_pre) ; stage in LDS
    float dpre = P[pbase + 256 + tid];
    float delta = (dpre > 20.0f) ? dpre : log1pf(__expf(dpre));
    sDelta[tid] = delta;

    // x_pool = mean of x_ssm (cols 0..127)
    float xs = P[pbase + tid];
    float ws = wave_sum32(xs);
    if ((tid & 31) == 0) sRed[tid >> 5] = ws;
    __syncthreads();
    const float x_pool = (sRed[0] + sRed[1] + sRed[2] + sRed[3]) * (1.0f / 128.0f);

    // Abar: 128 exps per thread; A column s read coalesced across lanes.
    const int s = tid;
    float acc = 0.0f;
#pragma unroll 4
    for (int i = 0; i < DINNER; ++i) {
        __builtin_prefetch(&A[(i + 16 < DINNER ? i + 16 : i) * DSTATE + s], 0, 1);
        acc += __expf(sDelta[i] * A[i * DSTATE + s]);
    }
    Abar[(size_t)row * DSTATE + s] = acc * (1.0f / 128.0f);
    Bx[(size_t)row * DSTATE + s]   = P[pbase + 384 + s] * x_pool;
}

// ----------------------------------------------------------------- K4: scan
__global__ void __launch_bounds__(128)
ssm_scan_kernel(const float* __restrict__ Abar,
                const float* __restrict__ Bx,
                const float* __restrict__ P,
                float* __restrict__ yscan) {
    __shared__ float red[2][4];
    const int b    = blockIdx.x;
    const int tid  = threadIdx.x;
    const int wave = tid >> 5;
    const int lane = tid & 31;

    float h = 0.0f;
    for (int l = 0; l < LL; ++l) {
        const size_t row = (size_t)b * LL + l;
        const float a  = Abar[row * DSTATE + tid];
        const float bx = Bx[row * DSTATE + tid];
        const float cc = P[row * NCOL + 512 + tid];
        h = fmaf(a, h, bx);
        float p = wave_sum32(cc * h);
        if (lane == 0) red[l & 1][wave] = p;
        __syncthreads();
        if (tid == 0) {
            yscan[row] = red[l & 1][0] + red[l & 1][1] +
                         red[l & 1][2] + red[l & 1][3];
        }
        // double-buffered red[] slots -> single barrier per step is enough
    }
}

// -------------------------------------------------- K5: gate + output GEMM
// outpre(4096x64) = (y*silu(gate))(4096x128) @ W_out^T(128x64), f16 WMMA K=128.
// W_out tile is async-DMA'd into LDS while the silu staging VALU work runs.
__global__ void __launch_bounds__(128)
ssm_outgemm_kernel(const float* __restrict__ P,
                   const float* __restrict__ yscan,
                   const _Float16* __restrict__ Wout_h,
                   float* __restrict__ outpre) {
    __shared__ __align__(16) _Float16 sA[64 * 128];   // 16 KB (activations)
    __shared__ __align__(16) _Float16 sB[64 * 128];   // 16 KB (W_out rows)

    const int tid = threadIdx.x;
    const int m0  = blockIdx.x * 64;

#if SSM_HAVE_ASYNC_LDS
    // kick off W_out tile DMA first; it overlaps the silu staging below
    {
        const uint4* gB = (const uint4*)Wout_h;
        uint4* lB = (uint4*)sB;
#pragma unroll
        for (int t = 0; t < 8; ++t)
            ssm_async_b128(gB + tid + 128 * t, lB + tid + 128 * t);
    }
#endif

    // stage y * silu(gate) as f16
    for (int idx = tid; idx < 64 * 128; idx += 128) {
        const int m = idx >> 7, i = idx & 127;
        const float g  = P[(size_t)(m0 + m) * NCOL + 128 + i];
        const float yv = yscan[m0 + m];
        const float sg = g / (1.0f + __expf(-g));    // silu
        sA[m * 128 + i] = (_Float16)(yv * sg);
    }

#if SSM_HAVE_ASYNC_LDS
    ssm_wait_async0();
    __syncthreads();
#else
    {
        const uint4* gB = (const uint4*)Wout_h;
        uint4* lB = (uint4*)sB;
#pragma unroll
        for (int t = 0; t < 8; ++t) lB[tid + 128 * t] = gB[tid + 128 * t];
    }
    __syncthreads();
#endif

    const int wave = tid >> 5;
    const int lane = tid & 31;
    const int ml   = lane & 15;
    const int hi   = lane >> 4;

    // 4 A fragments (k0 = 0,32,64,96) for this wave's 16 rows
    Frag16 afr[4];
    const int arow = (wave * 16 + ml) * 128;
#pragma unroll
    for (int kk = 0; kk < 4; ++kk) {
        const int k0 = kk * 32;
        afr[kk].q[0] = *(const uint4*)&sA[arow + k0 + hi * 8];
        afr[kk].q[1] = *(const uint4*)&sA[arow + k0 + 16 + hi * 8];
    }

#pragma unroll
    for (int ns = 0; ns < 4; ++ns) {
        const int ncol = ns * 16 + ml;       // 0..63 output feature
        v8f c = {};
#pragma unroll
        for (int kk = 0; kk < 4; ++kk) {
            const int k0 = kk * 32;
            Frag16 bfr;
            bfr.q[0] = *(const uint4*)&sB[ncol * 128 + k0 + hi * 8];
            bfr.q[1] = *(const uint4*)&sB[ncol * 128 + k0 + 16 + hi * 8];
            c = __builtin_amdgcn_wmma_f32_16x16x32_f16(false, afr[kk].h, false,
                                                       bfr.h, (short)0, c,
                                                       false, false);
        }
        const int gmbase = m0 + wave * 16 + hi * 8;
#pragma unroll
        for (int r = 0; r < 8; ++r)
            outpre[(size_t)(gmbase + r) * DMODEL + ncol] = c[r];
    }
}

// -------------------------------------------- K6: residual + layernorm(64)
__global__ void __launch_bounds__(256)
ssm_layernorm_kernel(const float* __restrict__ outpre,
                     const float* __restrict__ x,
                     const float* __restrict__ ln_w,
                     const float* __restrict__ ln_b,
                     float* __restrict__ out) {
    const int tid  = threadIdx.x;
    const int lane = tid & 31;
    const int row  = blockIdx.x * 8 + (tid >> 5);
    const size_t base = (size_t)row * DMODEL;

    float v0 = outpre[base + lane]      + x[base + lane];
    float v1 = outpre[base + lane + 32] + x[base + lane + 32];

    const float mu = wave_sum32(v0 + v1) * (1.0f / 64.0f);
    const float d0 = v0 - mu, d1 = v1 - mu;
    const float var = wave_sum32(d0 * d0 + d1 * d1) * (1.0f / 64.0f);
    const float rs  = rsqrtf(var + 1e-5f);

    out[base + lane]      = d0 * rs * ln_w[lane]      + ln_b[lane];
    out[base + lane + 32] = d1 * rs * ln_w[lane + 32] + ln_b[lane + 32];
}

// ---------------------------------------------------------------------------
static inline size_t align256(size_t v) { return (v + 255) & ~(size_t)255; }

extern "C" void kernel_launch(void* const* d_in, const int* in_sizes, int n_in,
                              void* d_out, int out_size, void* d_ws, size_t ws_size,
                              hipStream_t stream) {
    const float* x       = (const float*)d_in[0];
    const float* W_in    = (const float*)d_in[1];
    const float* W_delta = (const float*)d_in[2];
    const float* b_delta = (const float*)d_in[3];
    const float* W_B     = (const float*)d_in[4];
    const float* b_B     = (const float*)d_in[5];
    const float* W_C     = (const float*)d_in[6];
    const float* b_C     = (const float*)d_in[7];
    const float* A       = (const float*)d_in[8];
    const float* W_out   = (const float*)d_in[9];
    const float* ln_w    = (const float*)d_in[10];
    const float* ln_b    = (const float*)d_in[11];
    float* out = (float*)d_out;

    char* w = (char*)d_ws;
    _Float16* Xh     = (_Float16*)w;  w += align256((size_t)MROWS * DMODEL * 2);
    _Float16* Wall_h = (_Float16*)w;  w += align256((size_t)NCOL * DMODEL * 2);
    _Float16* Wout_h = (_Float16*)w;  w += align256((size_t)DMODEL * DINNER * 2);
    float* bias_all  = (float*)w;     w += align256((size_t)NCOL * 4);
    float* P         = (float*)w;     w += align256((size_t)MROWS * NCOL * 4);
    float* Abar      = (float*)w;     w += align256((size_t)MROWS * DSTATE * 4);
    float* Bx        = (float*)w;     w += align256((size_t)MROWS * DSTATE * 4);
    float* yscan     = (float*)w;     w += align256((size_t)MROWS * 4);
    float* outpre    = (float*)w;     w += align256((size_t)MROWS * DMODEL * 4);

    ssm_pack_kernel<<<(PACK_TOTAL + 255) / 256, 256, 0, stream>>>(
        x, W_in, W_delta, b_delta, W_B, b_B, W_C, b_C, W_out,
        Xh, Wall_h, Wout_h, bias_all);

    ssm_gemm_proj_kernel<<<dim3(MROWS / 64, NCOL / 64), 128, 0, stream>>>(
        Xh, Wall_h, bias_all, P);

    ssm_abar_kernel<<<MROWS, 128, 0, stream>>>(P, A, Abar, Bx);

    ssm_scan_kernel<<<BB, 128, 0, stream>>>(Abar, Bx, P, yscan);

    ssm_outgemm_kernel<<<MROWS / 64, 128, 0, stream>>>(P, yscan, Wout_h, outpre);

    ssm_layernorm_kernel<<<MROWS / 8, 256, 0, stream>>>(
        outpre, x, ln_w, ln_b, out);
}
